// GraphTransformer_24008867185162
// MI455X (gfx1250) — matrix operand order repaired and checked by reading, not compiled, not run on hardware
//
#include <hip/hip_runtime.h>
#include <hip/hip_bf16.h>

// ---------------------------------------------------------------------------
// CDNA5 (gfx1250) bf16-WMMA implementation of the 4-layer graph transformer.
// All GEMMs + both attention matmuls run on v_wmma_f32_16x16x32_bf16.
// Round 2: 32x64 per-wave GEMM tiles (2x B-fragment reuse), 32-row Q tiles in
// attention (2x K/V-fragment + V-staging reuse), loads grouped ahead of WMMAs.
// ---------------------------------------------------------------------------

typedef __bf16 bf16;
typedef __attribute__((ext_vector_type(16))) __bf16 v16bf;
typedef __attribute__((ext_vector_type(8)))  __bf16 v8bf;
typedef __attribute__((ext_vector_type(8)))  float  v8f;

#define BB   2
#define NN   2048
#define MTOT 4096          // B*N
#define DIM  640
#define NH   10
#define DH   64
#define DINNER 2560
#define FP   320           // FDIM=280 padded to 320
#define FDIM 280

static __device__ __forceinline__ v8f wmma_bf16(v16bf a, v16bf b, v8f c) {
  return __builtin_amdgcn_wmma_f32_16x16x32_bf16(false, a, false, b, (short)0, c,
                                                 false, false);
}

static __device__ __forceinline__ v16bf make_frag(const bf16* lo, const bf16* hi) {
  v8bf l = *(const v8bf*)lo;
  v8bf h = *(const v8bf*)hi;
  v16bf r;
#pragma unroll
  for (int i = 0; i < 8; ++i) { r[i] = l[i]; r[8 + i] = h[i]; }
  return r;
}

// A fragment (16x32 MxK) from row-major A (ld = leading dim in elements).
// Lane layout: lane m=lane&15, 16 contiguous K at k0=(lane>>4)*16.
static __device__ __forceinline__ v16bf load_a(const bf16* A, int row0, int ld,
                                               int kb, int lane) {
  int m  = lane & 15;
  int k0 = (lane >> 4) * 16;
  const bf16* p = A + (size_t)(row0 + m) * ld + kb + k0;
  return make_frag(p, p + 8);
}

// B fragment (32x16 KxN) where B[k][n] = W[col0+n][k], W row-major (N x K).
// Lane: n=lane&15, 16 contiguous K at kb + (lane>>4)*16.
static __device__ __forceinline__ v16bf load_b(const bf16* W, int col0, int ld,
                                               int kb, int lane) {
  int n  = lane & 15;
  int kk = kb + (lane >> 4) * 16;
  const bf16* p = W + (size_t)(col0 + n) * ld + kk;
  return make_frag(p, p + 8);
}

// ---------------------------------------------------------------------------
// Weight / bias conversion
// ---------------------------------------------------------------------------
__global__ void convert_w_kernel(const float* __restrict__ src, bf16* __restrict__ dst,
                                 int N, int K, int Np, int Kp) {
  int idx = blockIdx.x * blockDim.x + threadIdx.x;
  if (idx >= Np * Kp) return;
  int n = idx / Kp, k = idx - n * Kp;
  dst[idx] = (n < N && k < K) ? (bf16)src[(size_t)n * K + k] : (bf16)0.0f;
}

__global__ void pad_bias_kernel(const float* __restrict__ src, float* __restrict__ dst,
                                int N, int Np) {
  int idx = blockIdx.x * blockDim.x + threadIdx.x;
  if (idx < Np) dst[idx] = (idx < N) ? src[idx] : 0.0f;
}

// ---------------------------------------------------------------------------
// Embedding: Z[b,n,:] = [emb0[t0], emb1[t1], x_feats @ feat_w.T + feat_b]
// ---------------------------------------------------------------------------
__global__ void embed_kernel(const float* __restrict__ x_feats,
                             const int* __restrict__ x_toks,
                             const float* __restrict__ emb0,
                             const float* __restrict__ emb1,
                             const float* __restrict__ feat_w,
                             const float* __restrict__ feat_b,
                             float* __restrict__ Z) {
  int idx = blockIdx.x * blockDim.x + threadIdx.x;
  if (idx >= MTOT * DIM) return;
  int row = idx / DIM, c = idx - row * DIM;
  float v;
  if (c < 32) {
    int t0 = x_toks[row * 2 + 0];
    v = emb0[t0 * 32 + c];
  } else if (c < 64) {
    int t1 = x_toks[row * 2 + 1];
    v = emb1[t1 * 32 + (c - 32)];
  } else {
    int fc = c - 64;
    float acc = feat_b[fc];
    const float* xf = x_feats + (size_t)row * 19;
    const float* fw = feat_w + (size_t)fc * 19;
#pragma unroll
    for (int k = 0; k < 19; ++k) acc += xf[k] * fw[k];
    v = acc;
  }
  Z[idx] = v;
}

// ---------------------------------------------------------------------------
// LayerNorm over DIM=640, one wave per token, bf16 output.
// ---------------------------------------------------------------------------
__global__ void __launch_bounds__(128)
layernorm_kernel(const float* __restrict__ Z, const float* __restrict__ g,
                 const float* __restrict__ beta, bf16* __restrict__ out) {
  int lane = threadIdx.x & 31;
  int wv   = threadIdx.x >> 5;
  int row  = blockIdx.x * 4 + wv;
  const float* x = Z + (size_t)row * DIM;
  float vals[20];
  float s = 0.f, ss = 0.f;
#pragma unroll
  for (int i = 0; i < 20; ++i) {
    float v = x[lane + i * 32];
    vals[i] = v; s += v; ss += v * v;
  }
#pragma unroll
  for (int xm = 1; xm < 32; xm <<= 1) {
    s  += __shfl_xor(s, xm, 32);
    ss += __shfl_xor(ss, xm, 32);
  }
  float mean = s * (1.0f / DIM);
  float var  = ss * (1.0f / DIM) - mean * mean;
  float inv  = rsqrtf(var + 1e-5f);
  bf16* o = out + (size_t)row * DIM;
#pragma unroll
  for (int i = 0; i < 20; ++i) {
    int c = lane + i * 32;
    o[c] = (bf16)((vals[i] - mean) * inv * g[c] + beta[c]);
  }
}

// ---------------------------------------------------------------------------
// Generic bf16 WMMA GEMM: C = epi(A(MxK) @ W(NxK)^T + bias)
// Each wave: 32(M) x 64(N) tile; block = 4 waves stacked in M (128 rows).
// All 4 B fragments are loaded before the 8 WMMAs that consume them.
// ---------------------------------------------------------------------------
#define GF_BIAS 1
#define GF_RELU 2
#define GF_RES  4
#define GF_F32  8
#define GF_BF16 16
#define GF_QKV  32

__global__ void __launch_bounds__(128)
gemm_bf16_kernel(const bf16* __restrict__ A, const bf16* __restrict__ W,
                 const float* __restrict__ bias, const float* __restrict__ res,
                 float* __restrict__ Cf, bf16* __restrict__ Cb,
                 bf16* __restrict__ Vq, bf16* __restrict__ Qq, bf16* __restrict__ Kq,
                 int K, int ldc, int flags) {
  int lane = threadIdx.x & 31;
  int wv   = threadIdx.x >> 5;
  int row0 = (blockIdx.y * 4 + wv) * 32;
  int col0 = blockIdx.x * 64;

  v8f acc[2][4] = {};
  for (int kb = 0; kb < K; kb += 32) {
    v16bf a0 = load_a(A, row0,      K, kb, lane);
    v16bf a1 = load_a(A, row0 + 16, K, kb, lane);
    v16bf b[4];
#pragma unroll
    for (int t = 0; t < 4; ++t) b[t] = load_b(W, col0 + t * 16, K, kb, lane);
#pragma unroll
    for (int t = 0; t < 4; ++t) acc[0][t] = wmma_bf16(a0, b[t], acc[0][t]);
#pragma unroll
    for (int t = 0; t < 4; ++t) acc[1][t] = wmma_bf16(a1, b[t], acc[1][t]);
  }

  int nlo = lane & 15;
#pragma unroll
  for (int mh = 0; mh < 2; ++mh) {
    int rowbase = row0 + mh * 16 + ((lane >> 4) * 8);
#pragma unroll
    for (int t = 0; t < 4; ++t) {
      int col = col0 + t * 16 + nlo;
      float bv = (flags & GF_BIAS) ? bias[col] : 0.f;
#pragma unroll
      for (int r = 0; r < 8; ++r) {
        float v = acc[mh][t][r] + bv;
        if (flags & GF_RELU) v = v > 0.f ? v : 0.f;
        int row = rowbase + r;
        if (flags & GF_QKV) {
          int h  = col / 192;
          int rr = col - h * 192;
          int sel = rr >> 6, d = rr & 63;
          int bb  = row >> 11, tok = row & 2047;
          size_t off = ((size_t)(bb * NH + h) * NN + tok) * DH + d;
          bf16 bw = (bf16)v;
          if (sel == 0) Vq[off] = bw;
          else if (sel == 1) Qq[off] = bw;
          else Kq[off] = bw;
        } else {
          size_t off = (size_t)row * ldc + col;
          if (flags & GF_RES)  v += res[off];
          if (flags & GF_F32)  Cf[off] = v;
          if (flags & GF_BF16) Cb[off] = (bf16)v;
        }
      }
    }
  }
}

// ---------------------------------------------------------------------------
// Flash-style attention. One wave = one 32-row Q tile of one (b,h).
// Keys processed 32 at a time: S = Q K^T via WMMA (K frags shared across both
// 16-row halves), online softmax, P (via LDS relayout) x V (staged transposed
// in LDS once per key block, V frags shared across halves) via WMMA.
// Fuses mask add/mul, 1/sqrt(d), softmax, and leaky_relu epilogue.
// ---------------------------------------------------------------------------
__global__ void __launch_bounds__(128)
attention_kernel(const bf16* __restrict__ Qb, const bf16* __restrict__ Kb,
                 const bf16* __restrict__ Vb, const float* __restrict__ mask,
                 bf16* __restrict__ attn_out) {
  __shared__ __align__(32) bf16 Vt[4][DH * 32];   // V transposed: [dim][key]
  __shared__ __align__(32) bf16 Pl[4][32 * 32];   // P tile row-major, 32 rows

  int lane = threadIdx.x & 31;
  int wv   = threadIdx.x >> 5;
  int b    = blockIdx.z;
  int h    = blockIdx.y;
  int row0 = (blockIdx.x * 4 + wv) * 32;

  const size_t hb = (size_t)(b * NH + h) * NN * DH;
  const bf16* Qh = Qb + hb;
  const bf16* Kh = Kb + hb;
  const bf16* Vh = Vb + hb;
  const float* mg = mask + (size_t)b * NN;

  int nlo   = lane & 15;
  int khalf = lane >> 4;
  int rowb  = khalf * 8;

  v16bf qa[2][2];
#pragma unroll
  for (int hh = 0; hh < 2; ++hh) {
    qa[hh][0] = load_a(Qh, row0 + hh * 16, DH, 0, lane);
    qa[hh][1] = load_a(Qh, row0 + hh * 16, DH, 32, lane);
  }

  float maski[2][8], mr[2][8], lr[2][8];
#pragma unroll
  for (int hh = 0; hh < 2; ++hh)
#pragma unroll
    for (int r = 0; r < 8; ++r) {
      maski[hh][r] = mg[row0 + hh * 16 + rowb + r];
      mr[hh][r] = -1e30f;
      lr[hh][r] = 0.f;
    }
  v8f acc[2][4] = {};

  for (int kb = 0; kb < NN; kb += 32) {
    // ---- K fragments (shared by both row-halves) ----
    const bf16* k0p = Kh + (size_t)(kb + nlo) * DH + khalf * 16;
    const bf16* k1p = Kh + (size_t)(kb + 16 + nlo) * DH + khalf * 16;
    v16bf kf00 = make_frag(k0p, k0p + 8);
    v16bf kf01 = make_frag(k0p + 32, k0p + 40);
    v16bf kf10 = make_frag(k1p, k1p + 8);
    v16bf kf11 = make_frag(k1p + 32, k1p + 40);

    float mj0 = mg[kb + nlo];
    float mj1 = mg[kb + 16 + nlo];

    // ---- stage V transposed once per key block: Vt[dim][key], key = lane ----
    {
      const bf16* vrow = Vh + (size_t)(kb + lane) * DH;
      bf16* vt = Vt[wv];
#pragma unroll
      for (int d0 = 0; d0 < DH; d0 += 8) {
        v8bf vv = *(const v8bf*)(vrow + d0);
#pragma unroll
        for (int e = 0; e < 8; ++e) vt[(d0 + e) * 32 + lane] = vv[e];
      }
    }

#pragma unroll
    for (int hh = 0; hh < 2; ++hh) {
      // ---- S = Q K^T (two 16x16 key tiles) ----
      v8f s0 = {}, s1 = {};
      s0 = wmma_bf16(qa[hh][0], kf00, s0);
      s0 = wmma_bf16(qa[hh][1], kf01, s0);
      s1 = wmma_bf16(qa[hh][0], kf10, s1);
      s1 = wmma_bf16(qa[hh][1], kf11, s1);

      float p0[8], p1[8], t[8];
#pragma unroll
      for (int r = 0; r < 8; ++r) {
        float a0 = s0[r] * 0.125f + ((maski[hh][r] * mj0 > 0.f) ? 0.f : -1e20f);
        float a1 = s1[r] * 0.125f + ((maski[hh][r] * mj1 > 0.f) ? 0.f : -1e20f);
        p0[r] = a0; p1[r] = a1;
        t[r] = fmaxf(a0, a1);
      }
#pragma unroll
      for (int xm = 1; xm < 16; xm <<= 1)
#pragma unroll
        for (int r = 0; r < 8; ++r) t[r] = fmaxf(t[r], __shfl_xor(t[r], xm, 32));

      float scale[8];
#pragma unroll
      for (int r = 0; r < 8; ++r) {
        float nm = fmaxf(mr[hh][r], t[r]);
        scale[r] = __expf(mr[hh][r] - nm);
        mr[hh][r] = nm;
        p0[r] = __expf(p0[r] - nm);
        p1[r] = __expf(p1[r] - nm);
      }
      float rs[8];
#pragma unroll
      for (int r = 0; r < 8; ++r) rs[r] = p0[r] + p1[r];
#pragma unroll
      for (int xm = 1; xm < 16; xm <<= 1)
#pragma unroll
        for (int r = 0; r < 8; ++r) rs[r] += __shfl_xor(rs[r], xm, 32);
#pragma unroll
      for (int r = 0; r < 8; ++r) lr[hh][r] = lr[hh][r] * scale[r] + rs[r];
#pragma unroll
      for (int t4 = 0; t4 < 4; ++t4)
#pragma unroll
        for (int r = 0; r < 8; ++r) acc[hh][t4][r] *= scale[r];

      // ---- stage P (with mask_j applied) into LDS rows hh*16.. ----
      bf16* pl = Pl[wv] + hh * 16 * 32;
#pragma unroll
      for (int r = 0; r < 8; ++r) {
        pl[(rowb + r) * 32 + nlo]      = (bf16)(p0[r] * mj0);
        pl[(rowb + r) * 32 + 16 + nlo] = (bf16)(p1[r] * mj1);
      }
    }
    __builtin_amdgcn_wave_barrier();   // keep LDS stores before frag reads

    // ---- O += P x V (V frags shared by both halves) ----
    v16bf vf[4];
#pragma unroll
    for (int t4 = 0; t4 < 4; ++t4) {
      const bf16* vp = Vt[wv] + (t4 * 16 + nlo) * 32 + khalf * 16;
      vf[t4] = make_frag(vp, vp + 8);
    }
    int k0 = khalf * 16;
#pragma unroll
    for (int hh = 0; hh < 2; ++hh) {
      const bf16* pm = Pl[wv] + (hh * 16 + nlo) * 32 + k0;
      v16bf pa = make_frag(pm, pm + 8);
#pragma unroll
      for (int t4 = 0; t4 < 4; ++t4)
        acc[hh][t4] = wmma_bf16(pa, vf[t4], acc[hh][t4]);
    }
    __builtin_amdgcn_wave_barrier();
  }

  // ---- epilogue: * mask_i / l, leaky_relu, bf16 store ----
#pragma unroll
  for (int hh = 0; hh < 2; ++hh) {
    float fac[8];
#pragma unroll
    for (int r = 0; r < 8; ++r)
      fac[r] = lr[hh][r] > 0.f ? maski[hh][r] / lr[hh][r] : 0.f;
#pragma unroll
    for (int t4 = 0; t4 < 4; ++t4) {
#pragma unroll
      for (int r = 0; r < 8; ++r) {
        float v = acc[hh][t4][r] * fac[r];
        v = v > 0.f ? v : 0.01f * v;
        int row = row0 + hh * 16 + rowb + r;
        attn_out[((size_t)(b * NN + row)) * DIM + h * DH + t4 * 16 + nlo] = (bf16)v;
      }
    }
  }
}

// ---------------------------------------------------------------------------
// Final head dot: out[row] = h2[row, :280] . out_w + out_b
// ---------------------------------------------------------------------------
__global__ void final_dot_kernel(const float* __restrict__ h2,
                                 const float* __restrict__ out_w,
                                 const float* __restrict__ out_b,
                                 float* __restrict__ out) {
  int row = blockIdx.x * blockDim.x + threadIdx.x;
  if (row >= MTOT) return;
  const float* hp = h2 + (size_t)row * FP;
  float acc = out_b[0];
  for (int k = 0; k < FDIM; ++k) acc += hp[k] * out_w[k];
  out[row] = acc;
}

// ---------------------------------------------------------------------------
// Host launcher
// ---------------------------------------------------------------------------
extern "C" void kernel_launch(void* const* d_in, const int* in_sizes, int n_in,
                              void* d_out, int out_size, void* d_ws, size_t ws_size,
                              hipStream_t stream) {
  (void)in_sizes; (void)n_in; (void)out_size; (void)ws_size;

  const float* x_feats = (const float*)d_in[0];
  const int*   x_toks  = (const int*)d_in[1];
  const float* mask    = (const float*)d_in[2];
  const float* emb0    = (const float*)d_in[3];
  const float* emb1    = (const float*)d_in[4];
  const float* feat_w  = (const float*)d_in[5];
  const float* feat_b  = (const float*)d_in[6];
  const float* qkv_w   = (const float*)d_in[7];
  const float* qkv_b   = (const float*)d_in[8];
  const float* o_w     = (const float*)d_in[9];
  const float* ln1_g   = (const float*)d_in[10];
  const float* ln1_b   = (const float*)d_in[11];
  const float* ln2_g   = (const float*)d_in[12];
  const float* ln2_b   = (const float*)d_in[13];
  const float* p1_w    = (const float*)d_in[14];
  const float* p1_b    = (const float*)d_in[15];
  const float* p2_w    = (const float*)d_in[16];
  const float* p2_b    = (const float*)d_in[17];
  const float* fn_g    = (const float*)d_in[18];
  const float* fn_b    = (const float*)d_in[19];
  const float* lin1_w  = (const float*)d_in[20];
  const float* lin1_b  = (const float*)d_in[21];
  const float* rb1_w   = (const float*)d_in[22];
  const float* rb1_b   = (const float*)d_in[23];
  const float* rb2_w   = (const float*)d_in[24];
  const float* rb2_b   = (const float*)d_in[25];
  const float* out_w   = (const float*)d_in[26];
  const float* out_b   = (const float*)d_in[27];

  // ---- workspace carve ----
  size_t off = 0;
  auto alloc = [&](size_t bytes) -> char* {
    off = (off + 255) & ~(size_t)255;
    char* p = (char*)d_ws + off;
    off += bytes;
    return p;
  };
  bf16* qkvw_bf = (bf16*)alloc((size_t)4 * 1920 * 640 * 2);
  bf16* ow_bf   = (bf16*)alloc((size_t)4 * 640 * 640 * 2);
  bf16* p1w_bf  = (bf16*)alloc((size_t)4 * 2560 * 640 * 2);
  bf16* p2w_bf  = (bf16*)alloc((size_t)4 * 640 * 2560 * 2);
  bf16* l1w_bf  = (bf16*)alloc((size_t)FP * 640 * 2);
  bf16* r1w_bf  = (bf16*)alloc((size_t)FP * FP * 2);
  bf16* r2w_bf  = (bf16*)alloc((size_t)FP * FP * 2);
  float* l1b_p  = (float*)alloc(FP * 4);
  float* r1b_p  = (float*)alloc(FP * 4);
  float* r2b_p  = (float*)alloc(FP * 4);
  float* Z      = (float*)alloc((size_t)MTOT * DIM * 4);
  bf16* Zn_bf   = (bf16*)alloc((size_t)MTOT * DIM * 2);
  bf16* Qbuf    = (bf16*)alloc((size_t)BB * NH * NN * DH * 2);
  bf16* Kbuf    = (bf16*)alloc((size_t)BB * NH * NN * DH * 2);
  bf16* Vbuf    = (bf16*)alloc((size_t)BB * NH * NN * DH * 2);
  bf16* attn_bf = (bf16*)alloc((size_t)MTOT * DIM * 2);
  bf16* H_bf    = (bf16*)alloc((size_t)MTOT * DINNER * 2);
  float* h_f32  = (float*)alloc((size_t)MTOT * FP * 4);
  bf16* h_bf    = (bf16*)alloc((size_t)MTOT * FP * 2);
  bf16* r_bf    = (bf16*)alloc((size_t)MTOT * FP * 2);
  float* h2     = (float*)alloc((size_t)MTOT * FP * 4);

  auto conv = [&](const float* s, bf16* d, int N, int K, int Np, int Kp) {
    int total = Np * Kp;
    convert_w_kernel<<<(total + 255) / 256, 256, 0, stream>>>(s, d, N, K, Np, Kp);
  };
  conv(qkv_w, qkvw_bf, 4 * 1920, 640, 4 * 1920, 640);
  conv(o_w,   ow_bf,   4 * 640,  640, 4 * 640,  640);
  conv(p1_w,  p1w_bf,  4 * 2560, 640, 4 * 2560, 640);
  conv(p2_w,  p2w_bf,  4 * 640, 2560, 4 * 640, 2560);
  conv(lin1_w, l1w_bf, FDIM, 640, FP, 640);
  conv(rb1_w,  r1w_bf, FDIM, FDIM, FP, FP);
  conv(rb2_w,  r2w_bf, FDIM, FDIM, FP, FP);
  pad_bias_kernel<<<(FP + 255) / 256, 256, 0, stream>>>(lin1_b, l1b_p, FDIM, FP);
  pad_bias_kernel<<<(FP + 255) / 256, 256, 0, stream>>>(rb1_b, r1b_p, FDIM, FP);
  pad_bias_kernel<<<(FP + 255) / 256, 256, 0, stream>>>(rb2_b, r2b_p, FDIM, FP);

  embed_kernel<<<(MTOT * DIM + 255) / 256, 256, 0, stream>>>(
      x_feats, x_toks, emb0, emb1, feat_w, feat_b, Z);

  dim3 gblk(128);
  auto gemm = [&](const bf16* A, const bf16* W, const float* bias, const float* res,
                  float* Cf, bf16* Cb, int Np, int Kp, int flags) {
    dim3 grid(Np / 64, MTOT / 128);
    gemm_bf16_kernel<<<grid, gblk, 0, stream>>>(A, W, bias, res, Cf, Cb,
                                                Vbuf, Qbuf, Kbuf, Kp, Np, flags);
  };

  for (int l = 0; l < 4; ++l) {
    const bf16* qw = qkvw_bf + (size_t)l * 1920 * 640;
    const bf16* ow = ow_bf   + (size_t)l * 640 * 640;
    const bf16* w1 = p1w_bf  + (size_t)l * 2560 * 640;
    const bf16* w2 = p2w_bf  + (size_t)l * 640 * 2560;

    layernorm_kernel<<<MTOT / 4, 128, 0, stream>>>(Z, ln1_g + l * 640, ln1_b + l * 640, Zn_bf);
    gemm(Zn_bf, qw, qkv_b + l * 1920, nullptr, nullptr, nullptr,
         1920, 640, GF_BIAS | GF_QKV);
    attention_kernel<<<dim3(NN / 128, NH, BB), 128, 0, stream>>>(
        Qbuf, Kbuf, Vbuf, mask, attn_bf);
    gemm(attn_bf, ow, nullptr, Z, Z, nullptr, 640, 640, GF_RES | GF_F32);
    layernorm_kernel<<<MTOT / 4, 128, 0, stream>>>(Z, ln2_g + l * 640, ln2_b + l * 640, Zn_bf);
    gemm(Zn_bf, w1, p1_b + l * 2560, nullptr, nullptr, H_bf,
         2560, 640, GF_BIAS | GF_RELU | GF_BF16);
    gemm(H_bf, w2, p2_b + l * 640, Z, Z, nullptr,
         640, 2560, GF_BIAS | GF_RES | GF_F32);
  }

  layernorm_kernel<<<MTOT / 4, 128, 0, stream>>>(Z, fn_g, fn_b, Zn_bf);
  gemm(Zn_bf, l1w_bf, l1b_p, nullptr, h_f32, h_bf, FP, 640,
       GF_BIAS | GF_F32 | GF_BF16);
  gemm(h_bf, r1w_bf, r1b_p, nullptr, nullptr, r_bf, FP, FP,
       GF_BIAS | GF_RELU | GF_BF16);
  gemm(r_bf, r2w_bf, r2b_p, h_f32, h2, nullptr, FP, FP,
       GF_BIAS | GF_RES | GF_F32);

  final_dot_kernel<<<MTOT / 256, 256, 0, stream>>>(h2, out_w, out_b, (float*)d_out);
}
